// LinearAttention_64287070486842
// MI455X (gfx1250) — compile-verified, matrix operand order checked by reading, not run on hardware
//
#include <hip/hip_runtime.h>
#include <hip/hip_bf16.h>

// ---------------- CDNA5 WMMA types ----------------
typedef __attribute__((ext_vector_type(16))) __bf16 v16bf;
typedef __attribute__((ext_vector_type(8)))  float  v8f;

union FragBF16 {
  v16bf v;
  uint4 u[2];
};

#define WMMA_BF16(a, b, c) \
  __builtin_amdgcn_wmma_f32_16x16x32_bf16(false, (a), false, (b), (short)0, (c), false, false)

// ---------------- problem constants ----------------
constexpr int   kB     = 16;
constexpr int   kC     = 512;
constexpr int   kG     = 8;      // groups
constexpr int   kN     = 4096;   // H*W
constexpr int   kO     = 1536;   // 3C
constexpr int   kHeads = 8;
constexpr float kEps   = 1e-5f;

// =====================================================================
// 1) GroupNorm statistics: one block per (b, g); reduce 64*4096 floats.
//    stats[bg] = (mean, rsqrt(var+eps))
// =====================================================================
__global__ __launch_bounds__(256)
void gn_stats_kernel(const float* __restrict__ x, float2* __restrict__ stats,
                     long elemsPerGroup, float invCount) {
  const int bg = blockIdx.x;
  const float* p = x + (long)bg * elemsPerGroup;
  float s = 0.f, ss = 0.f;
  for (long i = threadIdx.x; i < elemsPerGroup; i += 256) {
    float v = p[i];
    s += v; ss += v * v;
  }
  __shared__ float rs[256], rss[256];
  rs[threadIdx.x] = s; rss[threadIdx.x] = ss;
  __syncthreads();
  for (int o = 128; o > 0; o >>= 1) {
    if ((int)threadIdx.x < o) {
      rs[threadIdx.x]  += rs[threadIdx.x + o];
      rss[threadIdx.x] += rss[threadIdx.x + o];
    }
    __syncthreads();
  }
  if (threadIdx.x == 0) {
    float mu  = rs[0] * invCount;
    float var = rss[0] * invCount - mu * mu;
    stats[bg] = make_float2(mu, rsqrtf(var + kEps));
  }
}

// =====================================================================
// 2) Fold GN1 into QKV weights (per batch):
//    xn[c] = a_c * x[c] + b_c  with a_c = rstd*gamma, b_c = beta - mu*a_c
//    qkv[o] = sum_c w[o,c]*a_c*x[c]  +  sum_c w[o,c]*b_c
//    One block per (b, o).
// =====================================================================
__global__ __launch_bounds__(256)
void fold_qkv_kernel(const float* __restrict__ w_qkv,
                     const float* __restrict__ gamma, const float* __restrict__ beta,
                     const float2* __restrict__ stats1,
                     __bf16* __restrict__ w1f, float* __restrict__ bias1) {
  const int bo = blockIdx.x;
  const int b = bo / kO, o = bo % kO;
  const float* wrow = w_qkv + (long)o * kC;
  float bacc = 0.f;
  for (int c = threadIdx.x; c < kC; c += 256) {
    float2 st = stats1[b * kG + (c >> 6)];      // C/G = 64
    float a = st.y * gamma[c];
    float w = wrow[c];
    w1f[((long)b * kO + o) * kC + c] = (__bf16)(w * a);
    bacc += w * (beta[c] - st.x * a);
  }
  __shared__ float rb[256];
  rb[threadIdx.x] = bacc;
  __syncthreads();
  for (int off = 128; off > 0; off >>= 1) {
    if ((int)threadIdx.x < off) rb[threadIdx.x] += rb[threadIdx.x + off];
    __syncthreads();
  }
  if (threadIdx.x == 0) bias1[(long)b * kO + o] = rb[0];
}

// =====================================================================
// 3) Convert w_out -> bf16
// =====================================================================
__global__ __launch_bounds__(256)
void cvt_f32_bf16_kernel(const float* __restrict__ in, __bf16* __restrict__ out, long n) {
  long i = (long)blockIdx.x * 256 + threadIdx.x;
  if (i < n) out[i] = (__bf16)in[i];
}

// =====================================================================
// 4) Shared WMMA GEMM:  Out[b, m, n] = sum_k A[b?, m, k] * B[b, k, n] + bias[b?, m]
//    BM=BN=128, BK=32, 256 threads = 8 waves (4 x 2), wave tile 32x64.
//    A is bf16 K-contiguous; B (f32 or bf16) transposed into LDS [n][k].
// =====================================================================
template <typename TB, bool OUT_F32>
__global__ __launch_bounds__(256)
void gemm_wmma_kernel(const __bf16* __restrict__ A, const TB* __restrict__ B,
                      const float* __restrict__ bias, void* __restrict__ Out,
                      int M, int N, int K,
                      long strideAb, long strideBb, long biasStride, long strideOb,
                      int tilesN) {
  __shared__ __align__(16) __bf16 sA[128 * 40];  // [m][k], stride 40 (pad)
  __shared__ __align__(16) __bf16 sB[128 * 40];  // [n][k], stride 40 (pad)

  const int tid  = threadIdx.x;
  const int wave = tid >> 5, lane = tid & 31;
  const int lrow = lane & 15, lhalf = lane >> 4;
  const int wm = wave & 3;   // 4 waves along M -> 32 rows each
  const int wn = wave >> 2;  // 2 waves along N -> 64 cols each
  const int tm = blockIdx.x / tilesN, tn = blockIdx.x % tilesN;
  const int bb = blockIdx.y;

  const __bf16* Ab = A + (long)bb * strideAb + (long)(tm * 128) * K;
  const TB*     Bb = B + (long)bb * strideBb + (long)(tn * 128);

  v8f acc[2][4];
#pragma unroll
  for (int i = 0; i < 2; ++i)
#pragma unroll
    for (int j = 0; j < 4; ++j) acc[i][j] = (v8f)0.f;

  for (int k0 = 0; k0 < K; k0 += 32) {
    // ---- stage A tile (128 x 32 bf16): each thread copies 16 contiguous ----
    {
      int e = tid * 16;
      int row = e >> 5, col = e & 31;
      const __bf16* src = Ab + (long)row * K + k0 + col;
      *(uint4*)&sA[row * 40 + col]     = *(const uint4*)(src);
      *(uint4*)&sA[row * 40 + col + 8] = *(const uint4*)(src + 8);
    }
    // ---- stage B tile (32 k-rows x 128 n), transpose into [n][k] ----
    {
      int kr = tid >> 3, n0 = (tid & 7) * 16;
      const TB* src = Bb + (long)(k0 + kr) * N + n0;
#pragma unroll
      for (int j = 0; j < 16; ++j)
        sB[(n0 + j) * 40 + kr] = (__bf16)((float)src[j]);
    }
    __syncthreads();

    // ---- fragments: lanes 0-15 hold K 0..7/16..23, lanes 16-31 hold 8..15/24..31 ----
    FragBF16 af[2], bf_[4];
#pragma unroll
    for (int i = 0; i < 2; ++i) {
      int m = wm * 32 + i * 16 + lrow;
      af[i].u[0] = *(const uint4*)&sA[m * 40 + lhalf * 8];
      af[i].u[1] = *(const uint4*)&sA[m * 40 + 16 + lhalf * 8];
    }
#pragma unroll
    for (int j = 0; j < 4; ++j) {
      int n = wn * 64 + j * 16 + lrow;
      bf_[j].u[0] = *(const uint4*)&sB[n * 40 + lhalf * 8];
      bf_[j].u[1] = *(const uint4*)&sB[n * 40 + 16 + lhalf * 8];
    }
#pragma unroll
    for (int i = 0; i < 2; ++i)
#pragma unroll
      for (int j = 0; j < 4; ++j)
        acc[i][j] = WMMA_BF16(af[i].v, bf_[j].v, acc[i][j]);
    __syncthreads();
  }

  // ---- epilogue: bias + store (C/D layout: VGPR r -> M = r + 8*lhalf, N = lrow) ----
  const float* biasb = bias + (long)bb * biasStride;
#pragma unroll
  for (int i = 0; i < 2; ++i)
#pragma unroll
    for (int j = 0; j < 4; ++j)
#pragma unroll
      for (int r = 0; r < 8; ++r) {
        int m = tm * 128 + wm * 32 + i * 16 + r + 8 * lhalf;
        int n = tn * 128 + wn * 64 + j * 16 + lrow;
        float v = acc[i][j][r] + biasb[m];
        if (OUT_F32)
          ((float*)Out)[(long)bb * strideOb + (long)m * N + n] = v;
        else
          ((__bf16*)Out)[(long)bb * strideOb + (long)m * N + n] = (__bf16)v;
      }
}

// =====================================================================
// 5) softmax over n (rows of 4096), in place on q region (channels 0..511)
// =====================================================================
__global__ __launch_bounds__(256)
void softmax_q_kernel(__bf16* __restrict__ qkv) {
  const int b = blockIdx.x >> 9, r = blockIdx.x & 511;
  __bf16* p = qkv + (long)b * kO * kN + (long)r * kN;
  __shared__ float red[256];
  float m = -3.0e38f;
  for (int i = threadIdx.x; i < kN; i += 256) m = fmaxf(m, (float)p[i]);
  red[threadIdx.x] = m; __syncthreads();
  for (int o = 128; o > 0; o >>= 1) {
    if ((int)threadIdx.x < o) red[threadIdx.x] = fmaxf(red[threadIdx.x], red[threadIdx.x + o]);
    __syncthreads();
  }
  m = red[0]; __syncthreads();
  float s = 0.f;
  for (int i = threadIdx.x; i < kN; i += 256) s += __expf((float)p[i] - m);
  red[threadIdx.x] = s; __syncthreads();
  for (int o = 128; o > 0; o >>= 1) {
    if ((int)threadIdx.x < o) red[threadIdx.x] += red[threadIdx.x + o];
    __syncthreads();
  }
  float inv = 1.f / red[0];
  for (int i = threadIdx.x; i < kN; i += 256)
    p[i] = (__bf16)(__expf((float)p[i] - m) * inv);
}

// =====================================================================
// 6) softmax over d (64, strided by n), in place on k region (chan 512..1023)
//    block = (b*8+h)*16 + chunk; thread owns one n (coalesced).
// =====================================================================
__global__ __launch_bounds__(256)
void softmax_k_kernel(__bf16* __restrict__ qkv) {
  const int chunk = blockIdx.x & 15, bh = blockIdx.x >> 4;
  const int b = bh >> 3, h = bh & 7;
  __bf16* base = qkv + (long)b * kO * kN + (long)(kC + h * 64) * kN + chunk * 256 + threadIdx.x;
  float m = -3.0e38f;
#pragma unroll 4
  for (int d = 0; d < 64; ++d) m = fmaxf(m, (float)base[(long)d * kN]);
  float s = 0.f;
#pragma unroll 4
  for (int d = 0; d < 64; ++d) s += __expf((float)base[(long)d * kN] - m);
  float inv = 1.f / s;
#pragma unroll 4
  for (int d = 0; d < 64; ++d)
    base[(long)d * kN] = (__bf16)(__expf((float)base[(long)d * kN] - m) * inv);
}

// =====================================================================
// 7) ctx[d,e] = sum_n k[d,n]*v[e,n]  per (b,h).  K=4096 split over 8 waves,
//    direct global fragments (both operands are n-contiguous), LDS reduce.
//    Epilogue stores ctxT bf16 [e][d] for the next GEMM's A operand.
// =====================================================================
__global__ __launch_bounds__(256)
void ctx_kernel(const __bf16* __restrict__ qkv, __bf16* __restrict__ ctxT) {
  const int bh = blockIdx.x, b = bh >> 3, h = bh & 7;
  const __bf16* k_ = qkv + (long)b * kO * kN + (long)(kC + h * 64) * kN;
  const __bf16* v_ = qkv + (long)b * kO * kN + (long)(2 * kC + h * 64) * kN;
  const int wave = threadIdx.x >> 5, lane = threadIdx.x & 31;
  const int lrow = lane & 15, lhalf = lane >> 4;

  v8f acc[4][4];
#pragma unroll
  for (int i = 0; i < 4; ++i)
#pragma unroll
    for (int j = 0; j < 4; ++j) acc[i][j] = (v8f)0.f;

  const int kbase = wave * 512;  // 4096 / 8 waves
  for (int k0 = 0; k0 < 512; k0 += 32) {
    FragBF16 af[4], bf_[4];
#pragma unroll
    for (int i = 0; i < 4; ++i) {
      const __bf16* ap = k_ + (long)(i * 16 + lrow) * kN + kbase + k0;
      af[i].u[0] = *(const uint4*)(ap + lhalf * 8);
      af[i].u[1] = *(const uint4*)(ap + 16 + lhalf * 8);
      const __bf16* bp = v_ + (long)(i * 16 + lrow) * kN + kbase + k0;
      bf_[i].u[0] = *(const uint4*)(bp + lhalf * 8);
      bf_[i].u[1] = *(const uint4*)(bp + 16 + lhalf * 8);
    }
#pragma unroll
    for (int i = 0; i < 4; ++i)
#pragma unroll
      for (int j = 0; j < 4; ++j)
        acc[i][j] = WMMA_BF16(af[i].v, bf_[j].v, acc[i][j]);
  }

  __shared__ float red[64 * 64];  // 16 KB
  for (int i = threadIdx.x; i < 4096; i += 256) red[i] = 0.f;
  __syncthreads();
#pragma unroll
  for (int i = 0; i < 4; ++i)
#pragma unroll
    for (int j = 0; j < 4; ++j)
#pragma unroll
      for (int r = 0; r < 8; ++r) {
        int d = i * 16 + r + 8 * lhalf;
        int e = j * 16 + lrow;
        atomicAdd(&red[d * 64 + e], acc[i][j][r]);
      }
  __syncthreads();
  __bf16* outp = ctxT + (long)bh * 4096;
  for (int i = threadIdx.x; i < 4096; i += 256) {
    int e = i >> 6, d = i & 63;
    outp[(long)e * 64 + d] = (__bf16)red[d * 64 + e];  // transposed store
  }
}

// =====================================================================
// 8) out[e,n] = sum_d ctxT[e,d] * q[d,n]  per (b,h,nblk).  K=64.
//    A direct from global; q tile transposed through LDS.
//    Output -> attn bf16 [b][c=h*64+e][n].
// =====================================================================
__global__ __launch_bounds__(256)
void attn_out_kernel(const __bf16* __restrict__ qkv, const __bf16* __restrict__ ctxT,
                     __bf16* __restrict__ attn) {
  const int nblk = blockIdx.x & 31, bh = blockIdx.x >> 5;
  const int b = bh >> 3, h = bh & 7;
  const __bf16* q_ = qkv + (long)b * kO * kN + (long)(h * 64) * kN + nblk * 128;
  const __bf16* cT = ctxT + (long)bh * 4096;  // [e][d]

  __shared__ __align__(16) __bf16 sQ[128 * 72];  // [n][d], stride 72 (pad)
  {
    int dr = threadIdx.x >> 2, n0 = (threadIdx.x & 3) * 32;
    const __bf16* src = q_ + (long)dr * kN + n0;
#pragma unroll
    for (int j = 0; j < 32; ++j) sQ[(n0 + j) * 72 + dr] = src[j];
  }
  __syncthreads();

  const int wave = threadIdx.x >> 5, lane = threadIdx.x & 31;
  const int lrow = lane & 15, lhalf = lane >> 4;
  const int wm = wave & 1;   // 2 waves along M(e) -> 32 each
  const int wn = wave >> 1;  // 4 waves along N(n) -> 32 each

  v8f acc[2][2];
#pragma unroll
  for (int i = 0; i < 2; ++i)
#pragma unroll
    for (int j = 0; j < 2; ++j) acc[i][j] = (v8f)0.f;

#pragma unroll
  for (int k0 = 0; k0 < 64; k0 += 32) {
    FragBF16 af[2], bf_[2];
#pragma unroll
    for (int i = 0; i < 2; ++i) {
      const __bf16* ap = cT + (long)(wm * 32 + i * 16 + lrow) * 64 + k0;
      af[i].u[0] = *(const uint4*)(ap + lhalf * 8);
      af[i].u[1] = *(const uint4*)(ap + 16 + lhalf * 8);
      int n = wn * 32 + i * 16 + lrow;
      bf_[i].u[0] = *(const uint4*)&sQ[n * 72 + k0 + lhalf * 8];
      bf_[i].u[1] = *(const uint4*)&sQ[n * 72 + k0 + 16 + lhalf * 8];
    }
#pragma unroll
    for (int i = 0; i < 2; ++i)
#pragma unroll
      for (int j = 0; j < 2; ++j)
        acc[i][j] = WMMA_BF16(af[i].v, bf_[j].v, acc[i][j]);
  }

  __bf16* outb = attn + (long)b * kC * kN + (long)(h * 64) * kN + nblk * 128;
#pragma unroll
  for (int i = 0; i < 2; ++i)
#pragma unroll
    for (int j = 0; j < 2; ++j)
#pragma unroll
      for (int r = 0; r < 8; ++r) {
        int e = wm * 32 + i * 16 + r + 8 * lhalf;
        int n = wn * 32 + j * 16 + lrow;
        outb[(long)e * kN + n] = (__bf16)acc[i][j][r];
      }
}

// =====================================================================
// 9) final: out = x + GN2(y)
// =====================================================================
__global__ __launch_bounds__(256)
void final_kernel(const float* __restrict__ x, const float* __restrict__ y,
                  const float2* __restrict__ stats2,
                  const float* __restrict__ gamma2, const float* __restrict__ beta2,
                  float* __restrict__ out) {
  long i = (long)blockIdx.x * 256 + threadIdx.x;
  int c = (int)((i >> 12) & (kC - 1));           // (i / 4096) % 512
  int bg = (int)(i >> 18);                       // i / (64*4096) = b*8+g
  float2 st = stats2[bg];
  float v = (y[i] - st.x) * st.y * gamma2[c] + beta2[c];
  out[i] = x[i] + v;
}

// =====================================================================
// launch
// =====================================================================
extern "C" void kernel_launch(void* const* d_in, const int* in_sizes, int n_in,
                              void* d_out, int out_size, void* d_ws, size_t ws_size,
                              hipStream_t stream) {
  const float* x     = (const float*)d_in[0];
  const float* gn1_w = (const float*)d_in[1];
  const float* gn1_b = (const float*)d_in[2];
  const float* w_qkv = (const float*)d_in[3];
  const float* w_out = (const float*)d_in[4];
  const float* b_out = (const float*)d_in[5];
  const float* gn2_w = (const float*)d_in[6];
  const float* gn2_b = (const float*)d_in[7];
  float* out = (float*)d_out;

  uint8_t* ws = (uint8_t*)d_ws;
  // workspace layout (bytes, 256-aligned)
  float2* stats1 = (float2*)(ws + 0);                         //   1 KB
  float2* stats2 = (float2*)(ws + 1024);                      //   1 KB
  float*  bias1  = (float*)(ws + 2048);                       //  96 KB
  __bf16* w1f    = (__bf16*)(ws + 100352);                    //  24 MB  [b][1536][512]
  __bf16* w2f    = (__bf16*)(ws + 25266176);                  // 0.5 MB  [512][512]
  __bf16* qkv    = (__bf16*)(ws + 25790464);                  // 192 MB  [b][1536][4096]
  __bf16* ctxT   = (__bf16*)(ws + 227117056);                 //   1 MB  [b*h][64][64]
  __bf16* attn   = (__bf16*)(ws + 228165632);                 //  64 MB  [b][512][4096]
  float*  y      = (float*)(ws + 295274496);                  // 128 MB  [b][512][4096]

  const long elemsPerGroup = (long)(kC / kG) * kN;  // 64*4096
  const float invCount = 1.0f / (float)elemsPerGroup;

  // 1) GN1 stats
  gn_stats_kernel<<<kB * kG, 256, 0, stream>>>(x, stats1, elemsPerGroup, invCount);
  // 2) fold GN1 into QKV weights (per batch) + bias
  fold_qkv_kernel<<<kB * kO, 256, 0, stream>>>(w_qkv, gn1_w, gn1_b, stats1, w1f, bias1);
  // 3) w_out -> bf16
  cvt_f32_bf16_kernel<<<(kC * kC + 255) / 256, 256, 0, stream>>>(w_out, w2f, (long)kC * kC);
  // 4) QKV GEMM: [1536,512] x [512,4096] per batch, bf16 WMMA, bf16 out
  gemm_wmma_kernel<float, false><<<dim3((kO / 128) * (kN / 128), kB), 256, 0, stream>>>(
      w1f, x, bias1, qkv, kO, kN, kC,
      (long)kO * kC, (long)kC * kN, (long)kO, (long)kO * kN, kN / 128);
  // 5) softmax q over n
  softmax_q_kernel<<<kB * kC, 256, 0, stream>>>(qkv);
  // 6) softmax k over d
  softmax_k_kernel<<<kB * kHeads * (kN / 256), 256, 0, stream>>>(qkv);
  // 7) ctx = k . v^T per (b,h)
  ctx_kernel<<<kB * kHeads, 256, 0, stream>>>(qkv, ctxT);
  // 8) out = ctx^T . q per (b,h)
  attn_out_kernel<<<kB * kHeads * (kN / 128), 256, 0, stream>>>(qkv, ctxT, attn);
  // 9) to_out GEMM: [512,512] x [512,4096] + bias, f32 out
  gemm_wmma_kernel<__bf16, true><<<dim3((kC / 128) * (kN / 128), kB), 256, 0, stream>>>(
      w2f, attn, b_out, y, kC, kN, kC,
      0L, (long)kC * kN, 0L, (long)kC * kN, kN / 128);
  // 10) GN2 stats on y
  gn_stats_kernel<<<kB * kG, 256, 0, stream>>>(y, stats2, elemsPerGroup, invCount);
  // 11) residual + GN2 apply
  final_kernel<<<(int)(((long)kB * kC * kN) / 256), 256, 0, stream>>>(
      x, y, stats2, gn2_w, gn2_b, out);
}
